// VectorQuantizer_52123723105069
// MI455X (gfx1250) — matrix operand order repaired and checked by reading, not compiled, not run on hardware
//
#include <hip/hip_runtime.h>
#include <hip/hip_bf16.h>

typedef __attribute__((ext_vector_type(16))) _Float16 v16h;
typedef __attribute__((ext_vector_type(8)))  float    v8f;

#define NUM_EMB   1024
#define EMB_DIM   64
#define HW        4096      // H*W
#define DHW       262144    // D*H*W
#define NVEC      262144    // B*H*W
#define NTILES    (NVEC / 16)        // 16384 row-tiles of 16
#define NPAIRS    (NTILES / 2)       // 8192 tile-pairs
#define TOTAL_OUT 16777216           // B*D*H*W

#define CB_BYTES   (NUM_EMB * EMB_DIM * 2)          // 128 KB f16 codebook
#define SMEM_BYTES (CB_BYTES + NUM_EMB * 4)         // + 4 KB e2

// ---------------------------------------------------------------------------
// Kernel 1: nearest-codebook-entry search with f16 WMMA (f32 accumulate).
// argmin_j ||x - e_j||^2  ==  argmin_j ( e2[j] - 2 * x.e_j )   (x2 constant)
// Each wave processes TWO 16-row x-tiles against each codebook B-tile:
// 4 WMMAs per 4 ds_load_b128 (2x arithmetic intensity vs round 1), with
// the next B-tile's loads issued before the current tile is consumed.
// ---------------------------------------------------------------------------
__global__ void vq_argmin_kernel(const float* __restrict__ latents,
                                 const float* __restrict__ codebook,
                                 int* __restrict__ ind) {
  extern __shared__ char smem[];
  _Float16* cb  = (_Float16*)smem;
  float*    e2s = (float*)(smem + CB_BYTES);

  const int tid = threadIdx.x;

  // Stage f16 codebook into LDS (coalesced) and exact f32 e2 rows.
  for (int j = tid; j < NUM_EMB * EMB_DIM; j += 256)
    cb[j] = (_Float16)codebook[j];
  for (int r = tid; r < NUM_EMB; r += 256) {
    const float* row = codebook + r * EMB_DIM;
    float s = 0.f;
#pragma unroll 8
    for (int d = 0; d < EMB_DIM; ++d) { float v = row[d]; s += v * v; }
    e2s[r] = s;
  }
  __syncthreads();

  const int lane = tid & 31;     // wave32
  const int li   = lane & 15;
  const int hi   = lane >> 4;
  const int wave   = (blockIdx.x * blockDim.x + tid) >> 5;
  const int nwaves = (gridDim.x * blockDim.x) >> 5;

  for (int tp = wave; tp < NPAIRS; tp += nwaves) {
    const int tile0 = tp * 2;

    // A operands for both tiles. k order used here (B uses the same): element i ->
    //   K = i + 16*hi; second operand covers K+32.
    v16h a0[2], a1[2];
#pragma unroll
    for (int t = 0; t < 2; ++t) {
      const int  n    = (tile0 + t) * 16 + li;      // this lane's A row
      const int  b    = n >> 12;                    // / 4096
      const long base = (long)b * DHW + (n & 4095); // + d*HW per element
#pragma unroll
      for (int i = 0; i < 16; ++i) {
        const int k = i + 16 * hi;
        a0[t][i] = (_Float16)latents[base + (long)k * HW];
        a1[t][i] = (_Float16)latents[base + (long)(k + 32) * HW];
      }
    }

    float minv[2][8];
    int   mini[2][8];
#pragma unroll
    for (int t = 0; t < 2; ++t)
#pragma unroll
      for (int r = 0; r < 8; ++r) { minv[t][r] = 3.4e38f; mini[t][r] = 0; }

    // B operand layout: lane holds column ct*16+li; element i -> K = i+16*hi.
    // Double-buffer: issue next tile's LDS loads before consuming current.
    v16h b0c = *(const v16h*)(cb + li * EMB_DIM + 16 * hi);
    v16h b1c = *(const v16h*)(cb + li * EMB_DIM + 32 + 16 * hi);

    for (int ct = 0; ct < NUM_EMB / 16; ++ct) {
      const int coln = ((ct + 1) & 63) * 16 + li;   // prefetch (wraps once, harmless)
      const v16h b0n = *(const v16h*)(cb + coln * EMB_DIM + 16 * hi);
      const v16h b1n = *(const v16h*)(cb + coln * EMB_DIM + 32 + 16 * hi);

      const int   col = ct * 16 + li;
      const float e2c = e2s[col];

      v8f acc[2];
#pragma unroll
      for (int t = 0; t < 2; ++t) {
        v8f c = {};
        c = __builtin_amdgcn_wmma_f32_16x16x32_f16(false, a0[t], false, b0c,
                                                   (short)0, c, false, false);
        c = __builtin_amdgcn_wmma_f32_16x16x32_f16(false, a1[t], false, b1c,
                                                   (short)0, c, false, false);
        acc[t] = c;
      }

#pragma unroll
      for (int t = 0; t < 2; ++t)
#pragma unroll
        for (int r = 0; r < 8; ++r) {
          // C/D layout: VGPR r, lanes 0-15 -> row r; lanes 16-31 -> row r+8.
          const float dv = e2c - 2.0f * acc[t][r];
          if (dv < minv[t][r]) { minv[t][r] = dv; mini[t][r] = col; }
        }

      b0c = b0n;
      b1c = b1n;
    }

    // Min+index tree across the 16 lanes of each half (columns live on lanes).
#pragma unroll
    for (int m = 1; m < 16; m <<= 1) {
#pragma unroll
      for (int t = 0; t < 2; ++t)
#pragma unroll
        for (int r = 0; r < 8; ++r) {
          const float ov = __shfl_xor(minv[t][r], m, 32);
          const int   oi = __shfl_xor(mini[t][r], m, 32);
          if (ov < minv[t][r] || (ov == minv[t][r] && oi < mini[t][r])) {
            minv[t][r] = ov; mini[t][r] = oi;
          }
        }
    }

    if (li == 0) {
#pragma unroll
      for (int t = 0; t < 2; ++t) {
        const int rowbase = (tile0 + t) * 16 + 8 * hi;
#pragma unroll
        for (int r = 0; r < 8; ++r) ind[rowbase + r] = mini[t][r];
      }
    }
  }
}

// ---------------------------------------------------------------------------
// Kernel 2: gather quantized output in [B,D,H,W] order + per-block loss sums.
// ---------------------------------------------------------------------------
#define K2_ELEMS 8
__global__ void vq_gather_loss_kernel(const float* __restrict__ latents,
                                      const float* __restrict__ codebook,
                                      const int* __restrict__ ind,
                                      float* __restrict__ outq,
                                      float* __restrict__ partial) {
  __shared__ float red[256];
  const int tid = threadIdx.x;
  const long base = (long)blockIdx.x * (256 * K2_ELEMS);

  float s = 0.f;
#pragma unroll
  for (int e = 0; e < K2_ELEMS; ++e) {
    const long i  = base + e * 256 + tid;           // flat [B,D,H,W] index
    const int  hw = (int)(i & 4095);
    const int  d  = (int)((i >> 12) & 63);
    const int  b  = (int)(i >> 18);
    const int  k  = ind[b * HW + hw];
    const float q = codebook[k * EMB_DIM + d];
    outq[i] = q;
    const float diff = latents[i] - q;
    s += diff * diff;
  }

  red[tid] = s;
  __syncthreads();
  for (int off = 128; off > 0; off >>= 1) {
    if (tid < off) red[tid] += red[tid + off];
    __syncthreads();
  }
  if (tid == 0) partial[blockIdx.x] = red[0];
}

// ---------------------------------------------------------------------------
// Kernel 3: deterministic final reduction -> vq_loss = 1.25 * MSE.
// ---------------------------------------------------------------------------
__global__ void vq_loss_finalize_kernel(const float* __restrict__ partial,
                                        int nparts, float* __restrict__ loss_out) {
  __shared__ float red[256];
  float s = 0.f;
  for (int j = threadIdx.x; j < nparts; j += 256) s += partial[j];
  red[threadIdx.x] = s;
  __syncthreads();
  for (int off = 128; off > 0; off >>= 1) {
    if (threadIdx.x < off) red[threadIdx.x] += red[threadIdx.x + off];
    __syncthreads();
  }
  if (threadIdx.x == 0)
    loss_out[0] = red[0] * (1.25f / (float)TOTAL_OUT);
}

// ---------------------------------------------------------------------------
extern "C" void kernel_launch(void* const* d_in, const int* in_sizes, int n_in,
                              void* d_out, int out_size, void* d_ws, size_t ws_size,
                              hipStream_t stream) {
  const float* latents  = (const float*)d_in[0];   // [64,64,64,64] f32
  const float* codebook = (const float*)d_in[1];   // [1024,64]     f32
  float* out = (float*)d_out;                      // [0]=loss, [1..]=quantized_st

  int*   ind     = (int*)d_ws;                               // 262144 ints
  float* partial = (float*)((char*)d_ws + NVEC * sizeof(int));

  // 512 blocks x 8 waves = 4096 waves -> 2 tile-pairs each; 128KB codebook/WG.
  vq_argmin_kernel<<<512, 256, SMEM_BYTES, stream>>>(latents, codebook, ind);

  const int nblocks2 = TOTAL_OUT / (256 * K2_ELEMS);         // 8192
  vq_gather_loss_kernel<<<nblocks2, 256, 0, stream>>>(latents, codebook, ind,
                                                      out + 1, partial);
  vq_loss_finalize_kernel<<<1, 256, 0, stream>>>(partial, nblocks2, out);
}
